// MolTransformerDual_53171695124482
// MI455X (gfx1250) — matrix operand order repaired and checked by reading, not compiled, not run on hardware
//
#include <hip/hip_runtime.h>

typedef __bf16 bf16;
typedef __attribute__((ext_vector_type(16))) __bf16 v16bf;
typedef __attribute__((ext_vector_type(8)))  float  v8f;
typedef int v4i __attribute__((ext_vector_type(4)));

static constexpr int Bsz = 256;
static constexpr int GN  = 64;
static constexpr int Nn  = Bsz * GN;     // 16384 nodes
static constexpr int Ee  = Nn * 4;       // 65536 edges
static constexpr int TT  = 32;

#define DEVFN static __device__ __forceinline__

// CDNA5 async global->LDS copy support (guarded; sync fallback keeps compile safe)
#if defined(__has_builtin)
#if __has_builtin(__builtin_amdgcn_global_load_async_to_lds_b128) && \
    __has_builtin(__builtin_amdgcn_s_wait_asynccnt)
#define HAVE_ASYNC_LDS 1
#endif
#endif
#ifndef HAVE_ASYNC_LDS
#define HAVE_ASYNC_LDS 0
#endif

#if HAVE_ASYNC_LDS
typedef __attribute__((address_space(1))) v4i gv4i;
typedef __attribute__((address_space(3))) v4i lv4i;
#endif

__constant__ int c_noff[9] = {0,119,128,139,151,160,165,173,175};
__constant__ int c_eoff[3] = {0,22,28};

// ---------------- WMMA fragment helpers (CDNA5 wave32 layouts) ----------------
DEVFN int laneId() { return (int)(threadIdx.x & 31u); }

// A matrix 16x32 bf16: lane&15 = M row; lanes>=16 carry K+8 interleave;
// vgpr v: k0 = (v>=4?16:0) + (v&3)*2
DEVFN v16bf loadFragA(const bf16* base, int stride) {
  int lane = laneId();
  int m  = lane & 15;
  int kb = (lane >= 16) ? 8 : 0;
  union { v16bf v; unsigned u[8]; } f;
  const bf16* p = base + m * stride + kb;
#pragma unroll
  for (int v2 = 0; v2 < 8; ++v2) {
    int k0 = ((v2 & 4) << 2) + ((v2 & 3) << 1);
    f.u[v2] = *(const unsigned*)(p + k0);
  }
  return f.v;
}

// B matrix 32x16 bf16, stored as Bt[n][k]: lane&15 = N col;
// lanes 0-15 K=0..7,16..23, lanes 16-31 K=8..15,24..31
DEVFN v16bf loadFragB(const bf16* base, int stride) {
  int lane = laneId();
  int n  = lane & 15;
  int kb = (lane >= 16) ? 8 : 0;
  union { v16bf v; unsigned u[8]; } f;
  const bf16* p = base + n * stride + kb;
#pragma unroll
  for (int v2 = 0; v2 < 8; ++v2)
    f.u[v2] = *(const unsigned*)(p + ((v2 & 4) << 2) + ((v2 & 3) << 1));
  return f.v;
}

DEVFN v8f wmma_bf16(v16bf a, v16bf b, v8f c) {
  return __builtin_amdgcn_wmma_f32_16x16x32_bf16(false, a, false, b, (short)0, c, false, false);
}

// ---------------- small utility kernels ----------------
__global__ void cvt_kernel(const float* __restrict__ s, bf16* __restrict__ d, int n) {
  int i = blockIdx.x * 256 + threadIdx.x;
  if (i < n) d[i] = (bf16)s[i];
}

__global__ void fill_u32_kernel(unsigned* __restrict__ p, unsigned v, int n) {
  int i = blockIdx.x * 256 + threadIdx.x;
  if (i < n) p[i] = v;
}

template<int NT>
DEVFN float blockSum(float v, float* red) {
  int t = (int)threadIdx.x;
  red[t] = v; __syncthreads();
#pragma unroll
  for (int s = NT >> 1; s > 0; s >>= 1) {
    if (t < s) red[t] += red[t + s];
    __syncthreads();
  }
  float r = red[0]; __syncthreads();
  return r;
}

// ---------------- embedding-gather fused GEMM: out[M,128] = gather(emb) @ W^T + b ----------------
template<int NF, bool OUT_BF16>
__global__ void embed_gemm_kernel(const int* __restrict__ idx, const bf16* __restrict__ emb,
                                  const bf16* __restrict__ W, const float* __restrict__ bias,
                                  void* __restrict__ outPtr) {
  const int K = NF * 128;
  __shared__ bf16 As[16 * 128];
  int tid = (int)threadIdx.x;
  int wave = tid >> 5, lane = tid & 31;
  v8f acc = {0.f,0.f,0.f,0.f,0.f,0.f,0.f,0.f};
  for (int f = 0; f < NF; ++f) {
    __syncthreads();
#pragma unroll
    for (int q = 0; q < 4; ++q) {
      int p = tid + q * 256;
      int r = p >> 6, c = (p & 63) << 1;
      int tr = (NF == 9 ? c_noff[f] : c_eoff[f]) + idx[(blockIdx.x * 16 + r) * NF + f];
      ((unsigned*)As)[p] = *(const unsigned*)(emb + (size_t)tr * 128 + c);
    }
    __syncthreads();
#pragma unroll
    for (int kk = 0; kk < 4; ++kk) {
      v16bf a = loadFragA(As + (kk << 5), 128);
      v16bf b = loadFragB(W + (size_t)(wave * 16) * K + (f << 7) + (kk << 5), K);
      acc = wmma_bf16(a, b, acc);
    }
  }
  int col = (wave << 4) + (lane & 15);
  float bv = bias[col];
  int rBase = blockIdx.x * 16 + ((lane >> 4) << 3);
#pragma unroll
  for (int v2 = 0; v2 < 8; ++v2) {
    float val = acc[v2] + bv;
    size_t o = (size_t)(rBase + v2) * 128 + col;
    if (OUT_BF16) ((bf16*)outPtr)[o] = (bf16)val;
    else          ((float*)outPtr)[o] = val;
  }
}

// ---------------- generic tiled WMMA GEMM: out[M,Nout] = A[M,K] @ W[Nout,K]^T + b ----------------
// K must be a multiple of 128.
template<bool A_BF16, bool RELU, bool OUT_BF16>
__global__ void gemm_kernel(const void* __restrict__ Aptr, int M, int K,
                            const bf16* __restrict__ W, const float* __restrict__ bias,
                            int Nout, void* __restrict__ outPtr) {
  __shared__ bf16 As[16 * 128];
  int tid = (int)threadIdx.x;
  int wave = tid >> 5, lane = tid & 31;
  int colTile = blockIdx.y * 8 + wave;
  bool active = (colTile * 16) < Nout;
  size_t rowBase = (size_t)blockIdx.x * 16;
  v8f acc = {0.f,0.f,0.f,0.f,0.f,0.f,0.f,0.f};
  int steps = K >> 7;
  for (int s = 0; s < steps; ++s) {
    __syncthreads();
    if (A_BF16) {
      const bf16* Ab = (const bf16*)Aptr;
#pragma unroll
      for (int q = 0; q < 4; ++q) {
        int p = tid + q * 256;
        int r = p >> 6, c = (p & 63) << 1;
        ((unsigned*)As)[p] = *(const unsigned*)(Ab + (rowBase + r) * K + (s << 7) + c);
      }
    } else {
      const float* Af = (const float*)Aptr;
#pragma unroll
      for (int q = 0; q < 4; ++q) {
        int p = tid + q * 256;
        int r = p >> 6, c = (p & 63) << 1;
        const float* sp = Af + (rowBase + r) * K + (s << 7) + c;
        As[(r << 7) + c]     = (bf16)sp[0];
        As[(r << 7) + c + 1] = (bf16)sp[1];
      }
    }
    __syncthreads();
    if (active) {
#pragma unroll
      for (int kk = 0; kk < 4; ++kk) {
        v16bf a = loadFragA(As + (kk << 5), 128);
        v16bf b = loadFragB(W + (size_t)(colTile * 16) * K + (s << 7) + (kk << 5), K);
        acc = wmma_bf16(a, b, acc);
      }
    }
  }
  if (active) {
    int col = (colTile << 4) + (lane & 15);
    float bv = bias ? bias[col] : 0.f;
    int rBase = blockIdx.x * 16 + ((lane >> 4) << 3);
#pragma unroll
    for (int v2 = 0; v2 < 8; ++v2) {
      float val = acc[v2] + bv;
      if (RELU) val = fmaxf(val, 0.f);
      size_t o = (size_t)(rBase + v2) * Nout + col;
      if (OUT_BF16) ((bf16*)outPtr)[o] = (bf16)val;
      else          ((float*)outPtr)[o] = val;
    }
  }
}

// ---------------- edge attention (irregular, VALU + atomics) ----------------
DEVFN unsigned encodeF(float f) {
  unsigned u = __float_as_uint(f);
  return (u >> 31) ? ~u : (u | 0x80000000u);
}
DEVFN float decodeF(unsigned k) {
  unsigned u = (k >> 31) ? (k ^ 0x80000000u) : ~k;
  return __uint_as_float(u);
}

__global__ void attn_score_kernel(const float* __restrict__ q, const float* __restrict__ k,
                                  const float* __restrict__ elin,
                                  const int* __restrict__ src, const int* __restrict__ dst,
                                  float* __restrict__ aS, unsigned* __restrict__ amax) {
  int gid = blockIdx.x * 256 + threadIdx.x;
  int e = gid >> 2, h = gid & 3;
  int s = src[e], d = dst[e];
  const float4* qp = (const float4*)(q + (size_t)d * 128 + h * 32);
  const float4* kp = (const float4*)(k + (size_t)s * 128 + h * 32);
  const float4* ep = (const float4*)(elin + (size_t)e * 128 + h * 32);
  float acc = 0.f;
#pragma unroll
  for (int i = 0; i < 8; ++i) {
    float4 qv = qp[i], kv = kp[i], ev = ep[i];
    acc += qv.x * (kv.x + ev.x) + qv.y * (kv.y + ev.y)
         + qv.z * (kv.z + ev.z) + qv.w * (kv.w + ev.w);
  }
  acc *= 0.17677669529663687f; // 1/sqrt(32)
  aS[gid] = acc;
  atomicMax(&amax[d * 4 + h], encodeF(acc));
}

__global__ void attn_norm_kernel(float* __restrict__ aS, const unsigned* __restrict__ amax,
                                 const int* __restrict__ dst, float* __restrict__ den) {
  int gid = blockIdx.x * 256 + threadIdx.x;
  int e = gid >> 2, h = gid & 3;
  int d = dst[e];
  float m = decodeF(amax[d * 4 + h]);
  float ae = __expf(aS[gid] - m);
  aS[gid] = ae;
  atomicAdd(&den[d * 4 + h], ae);
}

__global__ void attn_agg_kernel(const float* __restrict__ ae, const float* __restrict__ den,
                                const float* __restrict__ v, const float* __restrict__ elin,
                                const int* __restrict__ src, const int* __restrict__ dst,
                                float* __restrict__ agg) {
  int gid = blockIdx.x * 256 + threadIdx.x;
  int e = gid >> 5;
  int q5 = gid & 31;
  int h = q5 >> 3, c0 = (q5 & 7) << 2;
  int s = src[e], d = dst[e];
  float alpha = ae[e * 4 + h] / den[d * 4 + h];
  int ch = h * 32 + c0;
  float4 vv = *(const float4*)(v + (size_t)s * 128 + ch);
  float4 ev = *(const float4*)(elin + (size_t)e * 128 + ch);
  float* ap = agg + (size_t)d * 128 + ch;
  atomicAdd(ap + 0, (vv.x + ev.x) * alpha);
  atomicAdd(ap + 1, (vv.y + ev.y) * alpha);
  atomicAdd(ap + 2, (vv.z + ev.z) * alpha);
  atomicAdd(ap + 3, (vv.w + ev.w) * alpha);
}

// beta gate + residual + LayerNorm + ReLU, in-place on x. One node per 128-thread block.
__global__ void beta_ln_kernel(const float* __restrict__ agg, const float* __restrict__ skip,
                               float* __restrict__ x, const float* __restrict__ betaW,
                               const float* __restrict__ g, const float* __restrict__ b) {
  __shared__ float red[128];
  int n = blockIdx.x, t = (int)threadIdx.x;
  float o = agg[(size_t)n * 128 + t];
  float r = skip[(size_t)n * 128 + t];
  float xin = x[(size_t)n * 128 + t];
  float part = betaW[t] * o + betaW[128 + t] * r + betaW[256 + t] * (o - r);
  float s = blockSum<128>(part, red);
  float beta = 1.f / (1.f + __expf(-s));
  float y = xin + beta * r + (1.f - beta) * o;
  float mu = blockSum<128>(y, red) * (1.f / 128.f);
  float d = y - mu;
  float var = blockSum<128>(d * d, red) * (1.f / 128.f);
  float z = d * rsqrtf(var + 1e-5f) * g[t] + b[t];
  x[(size_t)n * 128 + t] = fmaxf(z, 0.f);
}

// LN(192) + L2 normalize + bf16 store. 64 threads/row, 3 channels each.
__global__ void ln_l2_kernel(const float* __restrict__ in, const float* __restrict__ g,
                             const float* __restrict__ b, bf16* __restrict__ out) {
  __shared__ float red[64];
  int n = blockIdx.x, t = (int)threadIdx.x;
  const float* row = in + (size_t)n * 192;
  float y0 = row[t], y1 = row[t + 64], y2 = row[t + 128];
  float mu = blockSum<64>(y0 + y1 + y2, red) * (1.f / 192.f);
  float d0 = y0 - mu, d1 = y1 - mu, d2 = y2 - mu;
  float var = blockSum<64>(d0 * d0 + d1 * d1 + d2 * d2, red) * (1.f / 192.f);
  float is = rsqrtf(var + 1e-5f);
  float z0 = d0 * is * g[t] + b[t];
  float z1 = d1 * is * g[t + 64] + b[t + 64];
  float z2 = d2 * is * g[t + 128] + b[t + 128];
  float nrm = sqrtf(blockSum<64>(z0 * z0 + z1 * z1 + z2 * z2, red));
  float inv = 1.f / fmaxf(nrm, 1e-12f);
  bf16* orow = out + (size_t)n * 192;
  orow[t] = (bf16)(z0 * inv); orow[t + 64] = (bf16)(z1 * inv); orow[t + 128] = (bf16)(z2 * inv);
}

// L2 normalize (192) + bf16 store (text tokens)
__global__ void l2_kernel(const float* __restrict__ in, bf16* __restrict__ out) {
  __shared__ float red[64];
  int n = blockIdx.x, t = (int)threadIdx.x;
  const float* row = in + (size_t)n * 192;
  float z0 = row[t], z1 = row[t + 64], z2 = row[t + 128];
  float nrm = sqrtf(blockSum<64>(z0 * z0 + z1 * z1 + z2 * z2, red));
  float inv = 1.f / fmaxf(nrm, 1e-12f);
  bf16* orow = out + (size_t)n * 192;
  orow[t] = (bf16)(z0 * inv); orow[t + 64] = (bf16)(z1 * inv); orow[t + 128] = (bf16)(z2 * inv);
}

// ---------------- ColBERT MaxSim scoring ----------------
// out[i,j] = sum_t max_g <T_i[t], G_j[g]>, double-buffered G staging (async-to-LDS if available)
DEVFN void stageG(const bf16* __restrict__ src, bf16* dst, int tid) {
#if HAVE_ASYNC_LDS
  // 64*192 bf16 = 24576B = 1536 x b128 lane-transfers; 256 threads x 6
#pragma unroll
  for (int q = 0; q < 6; ++q) {
    int p = tid + q * 256;
    __builtin_amdgcn_global_load_async_to_lds_b128(
        (gv4i*)(src + p * 8), (lv4i*)(dst + p * 8), 0, 0);
  }
#else
  const unsigned* sp = (const unsigned*)src;
  unsigned* dp = (unsigned*)dst;
#pragma unroll
  for (int q = 0; q < 24; ++q) {
    int p = tid + q * 256;
    dp[p] = sp[p];
  }
#endif
}

__global__ void score_kernel(const bf16* __restrict__ ttok, const bf16* __restrict__ gtok,
                             float* __restrict__ out) {
  __shared__ bf16 Ts[32 * 192];
  __shared__ bf16 Gs[2][64 * 192];
  __shared__ float Ss[32 * 64];
  __shared__ float rmax[32];
  int i = blockIdx.x, jb = blockIdx.y;
  int tid = (int)threadIdx.x, wave = tid >> 5, lane = tid & 31;
  int mt = wave >> 2, nt = wave & 3;
  {
    const unsigned* sp = (const unsigned*)(ttok + (size_t)i * (32 * 192));
    unsigned* dp = (unsigned*)Ts;
#pragma unroll
    for (int q = 0; q < 12; ++q) dp[tid + q * 256] = sp[tid + q * 256];
  }
  int jb8 = jb * 8;
  stageG(gtok + (size_t)jb8 * (64 * 192), Gs[0], tid);
  for (int jj = 0; jj < 8; ++jj) {
#if HAVE_ASYNC_LDS
    __builtin_amdgcn_s_wait_asynccnt(0);
#endif
    __syncthreads();  // all waves' G data landed; previous buffer's readers done
    if (jj + 1 < 8) stageG(gtok + (size_t)(jb8 + jj + 1) * (64 * 192), Gs[(jj + 1) & 1], tid);
    const bf16* G = Gs[jj & 1];
    v8f acc = {0.f,0.f,0.f,0.f,0.f,0.f,0.f,0.f};
#pragma unroll
    for (int kk = 0; kk < 6; ++kk) {
      v16bf a = loadFragA(Ts + mt * 16 * 192 + (kk << 5), 192);
      v16bf b = loadFragB(G + nt * 16 * 192 + (kk << 5), 192);
      acc = wmma_bf16(a, b, acc);
    }
    int col = lane & 15, rb = (lane >> 4) << 3;
#pragma unroll
    for (int v2 = 0; v2 < 8; ++v2)
      Ss[(mt * 16 + rb + v2) * 64 + nt * 16 + col] = acc[v2];
    __syncthreads();
    if (tid < 32) {
      float m = -3.4e38f;
      for (int c = 0; c < 64; ++c) m = fmaxf(m, Ss[tid * 64 + c]);
      rmax[tid] = m;
    }
    __syncthreads();
    if (tid == 0) {
      float s = 0.f;
      for (int r = 0; r < 32; ++r) s += rmax[r];
      out[(size_t)i * 256 + (jb8 + jj)] = s;
    }
  }
}

// ---------------- host orchestration ----------------
extern "C" void kernel_launch(void* const* d_in, const int* in_sizes, int n_in,
                              void* d_out, int out_size, void* d_ws, size_t ws_size,
                              hipStream_t stream) {
  (void)in_sizes; (void)n_in; (void)out_size; (void)ws_size;

  const int* x_nodes    = (const int*)d_in[0];
  const int* ei_src     = (const int*)d_in[1];
  const int* ei_dst     = ei_src + Ee;
  const int* edge_attr  = (const int*)d_in[2];
  const float* text_emb = (const float*)d_in[4];
  auto F = [&](int i) { return (const float*)d_in[i]; };

  // ---- workspace layout ----
  size_t off = 0;
  auto take = [&](size_t bytes) -> size_t {
    size_t o = off; off += (bytes + 255) & ~(size_t)255; return o;
  };
  char* wsp = (char*)d_ws;
  bf16* nodeEmbBf = (bf16*)(wsp + take((size_t)177 * 128 * 2));
  bf16* WnpBf     = (bf16*)(wsp + take((size_t)128 * 1152 * 2));
  bf16* edgeEmbBf = (bf16*)(wsp + take((size_t)30 * 128 * 2));
  bf16* WepBf     = (bf16*)(wsp + take((size_t)128 * 384 * 2));
  bf16* convWBf   = (bf16*)(wsp + take((size_t)15 * 128 * 128 * 2));
  bf16* WgBf      = (bf16*)(wsp + take((size_t)192 * 128 * 2));
  bf16* Wl1Bf     = (bf16*)(wsp + take((size_t)256 * 768 * 2));
  bf16* Wl2Bf     = (bf16*)(wsp + take((size_t)6144 * 256 * 2));
  float* xbuf     = (float*)(wsp + take((size_t)Nn * 128 * 4));
  bf16* eaBf      = (bf16*)(wsp + take((size_t)Ee * 128 * 2));
  float* qb       = (float*)(wsp + take((size_t)Nn * 128 * 4));
  float* kb       = (float*)(wsp + take((size_t)Nn * 128 * 4));
  float* vb       = (float*)(wsp + take((size_t)Nn * 128 * 4));
  float* sb       = (float*)(wsp + take((size_t)Nn * 128 * 4));
  float* elin     = (float*)(wsp + take((size_t)Ee * 128 * 4));
  float* aS       = (float*)(wsp + take((size_t)Ee * 4 * 4));
  unsigned* amax  = (unsigned*)(wsp + take((size_t)Nn * 4 * 4));
  float* den      = (float*)(wsp + take((size_t)Nn * 4 * 4));
  float* agg      = (float*)(wsp + take((size_t)Nn * 128 * 4));
  float* tokpre   = (float*)(wsp + take((size_t)Nn * 192 * 4));
  bf16* tokBf     = (bf16*)(wsp + take((size_t)Nn * 192 * 2));
  bf16* hBf       = (bf16*)(wsp + take((size_t)256 * 256 * 2));
  float* ttokpre  = (float*)(wsp + take((size_t)256 * 6144 * 4));
  bf16* ttokBf    = (bf16*)(wsp + take((size_t)Bsz * TT * 192 * 2));

  auto cvt = [&](const void* src, bf16* dst, int n) {
    cvt_kernel<<<(n + 255) / 256, 256, 0, stream>>>((const float*)src, dst, n);
  };

  // ---- weight conversions (f32 -> bf16) ----
  static const int XS[9]   = {119,9,11,12,9,5,8,2,2};
  static const int NOFF[9] = {0,119,128,139,151,160,165,173,175};
  static const int ES[3]   = {22,6,2};
  static const int EOFF[3] = {0,22,28};
  for (int f = 0; f < 9; ++f) cvt(d_in[5 + f], nodeEmbBf + (size_t)NOFF[f] * 128, XS[f] * 128);
  cvt(d_in[14], WnpBf, 128 * 1152);
  for (int f = 0; f < 3; ++f) cvt(d_in[16 + f], edgeEmbBf + (size_t)EOFF[f] * 128, ES[f] * 128);
  cvt(d_in[19], WepBf, 128 * 384);
  for (int L = 0; L < 3; ++L) {
    int CL = 21 + L * 10;
    cvt(d_in[CL + 0], convWBf + (size_t)(L * 5 + 0) * 16384, 128 * 128); // q
    cvt(d_in[CL + 2], convWBf + (size_t)(L * 5 + 1) * 16384, 128 * 128); // k
    cvt(d_in[CL + 4], convWBf + (size_t)(L * 5 + 2) * 16384, 128 * 128); // v
    cvt(d_in[CL + 7], convWBf + (size_t)(L * 5 + 3) * 16384, 128 * 128); // skip
    cvt(d_in[CL + 6], convWBf + (size_t)(L * 5 + 4) * 16384, 128 * 128); // e
  }
  cvt(d_in[57], WgBf, 192 * 128);
  cvt(d_in[61], Wl1Bf, 256 * 768);
  cvt(d_in[63], Wl2Bf, 6144 * 256);

  // ---- embeddings + input projections ----
  embed_gemm_kernel<9, false><<<Nn / 16, 256, 0, stream>>>(x_nodes, nodeEmbBf, WnpBf, F(15), xbuf);
  embed_gemm_kernel<3, true><<<Ee / 16, 256, 0, stream>>>(edge_attr, edgeEmbBf, WepBf, F(20), eaBf);

  // ---- 3 TransformerConv layers ----
  for (int L = 0; L < 3; ++L) {
    int CL = 21 + L * 10;
    const bf16* Wq = convWBf + (size_t)(L * 5 + 0) * 16384;
    const bf16* Wk = convWBf + (size_t)(L * 5 + 1) * 16384;
    const bf16* Wv = convWBf + (size_t)(L * 5 + 2) * 16384;
    const bf16* Ws = convWBf + (size_t)(L * 5 + 3) * 16384;
    const bf16* We = convWBf + (size_t)(L * 5 + 4) * 16384;
    gemm_kernel<false, false, false><<<dim3(Nn / 16, 1), 256, 0, stream>>>(xbuf, Nn, 128, Wq, F(CL + 1), 128, qb);
    gemm_kernel<false, false, false><<<dim3(Nn / 16, 1), 256, 0, stream>>>(xbuf, Nn, 128, Wk, F(CL + 3), 128, kb);
    gemm_kernel<false, false, false><<<dim3(Nn / 16, 1), 256, 0, stream>>>(xbuf, Nn, 128, Wv, F(CL + 5), 128, vb);
    gemm_kernel<false, false, false><<<dim3(Nn / 16, 1), 256, 0, stream>>>(xbuf, Nn, 128, Ws, F(CL + 8), 128, sb);
    gemm_kernel<true, false, false><<<dim3(Ee / 16, 1), 256, 0, stream>>>(eaBf, Ee, 128, We, nullptr, 128, elin);

    fill_u32_kernel<<<(Nn * 4 + 255) / 256, 256, 0, stream>>>(amax, 0x007FFFFFu, Nn * 4);
    (void)hipMemsetAsync(den, 0, (size_t)Nn * 4 * sizeof(float), stream);
    (void)hipMemsetAsync(agg, 0, (size_t)Nn * 128 * sizeof(float), stream);

    attn_score_kernel<<<Ee * 4 / 256, 256, 0, stream>>>(qb, kb, elin, ei_src, ei_dst, aS, amax);
    attn_norm_kernel<<<Ee * 4 / 256, 256, 0, stream>>>(aS, amax, ei_dst, den);
    attn_agg_kernel<<<Ee * 32 / 256, 256, 0, stream>>>(aS, den, vb, elin, ei_src, ei_dst, agg);
    beta_ln_kernel<<<Nn, 128, 0, stream>>>(agg, sb, xbuf, F(CL + 9), F(51 + 2 * L), F(52 + 2 * L));
  }

  // ---- graph projection + LN + L2 -> bf16 tokens ----
  gemm_kernel<false, false, false><<<dim3(Nn / 16, 2), 256, 0, stream>>>(xbuf, Nn, 128, WgBf, F(58), 192, tokpre);
  ln_l2_kernel<<<Nn, 64, 0, stream>>>(tokpre, F(59), F(60), tokBf);

  // ---- text tower ----
  gemm_kernel<false, true, true><<<dim3(256 / 16, 2), 256, 0, stream>>>(text_emb, 256, 768, Wl1Bf, F(62), 256, hBf);
  gemm_kernel<true, false, false><<<dim3(256 / 16, 48), 256, 0, stream>>>(hBf, 256, 256, Wl2Bf, F(64), 6144, ttokpre);
  l2_kernel<<<Bsz * TT, 64, 0, stream>>>(ttokpre, ttokBf);

  // ---- ColBERT MaxSim scoring ----
  score_kernel<<<dim3(256, 32), 256, 0, stream>>>(ttokBf, tokBf, (float*)d_out);
}
